// Mamba_36404142801309
// MI455X (gfx1250) — compile-verified, matrix operand order checked by reading, not run on hardware
//
#include <hip/hip_runtime.h>
#include <hip/hip_bf16.h>
#include <math.h>

#define D_MODEL 768
#define N_LAYERS 2
#define VOCAB   50264
#define D_STATE 16
#define D_INNER 1536
#define DT_RANK 48
#define DT_PAD  64                 // DT_RANK padded to multiple of 32
#define D_CONV  4
#define BATCH   2
#define SEQ     1024
#define NTOK    (BATCH * SEQ)
#define XPROJ_N (DT_RANK + 2 * D_STATE)   // 80

typedef __attribute__((ext_vector_type(16))) __bf16 v16bf;
typedef __attribute__((ext_vector_type(8)))  __bf16 v8bf;
typedef __attribute__((ext_vector_type(8)))  float  v8f;

__device__ __forceinline__ float silu_f(float x) {
  return x / (1.0f + expf(-x));
}
__device__ __forceinline__ float softplus_f(float x) {
  return (x > 20.0f) ? x : log1pf(expf(x));
}
__device__ __forceinline__ v16bf cat16(v8bf a, v8bf b) {
  return __builtin_shufflevector(a, b, 0, 1, 2, 3, 4, 5, 6, 7,
                                 8, 9, 10, 11, 12, 13, 14, 15);
}
__device__ __forceinline__ v16bf cvt16(float4 f0, float4 f1, float4 f2, float4 f3) {
  v16bf r;
  r[0] = (__bf16)f0.x;  r[1] = (__bf16)f0.y;  r[2] = (__bf16)f0.z;  r[3] = (__bf16)f0.w;
  r[4] = (__bf16)f1.x;  r[5] = (__bf16)f1.y;  r[6] = (__bf16)f1.z;  r[7] = (__bf16)f1.w;
  r[8] = (__bf16)f2.x;  r[9] = (__bf16)f2.y;  r[10] = (__bf16)f2.z; r[11] = (__bf16)f2.w;
  r[12] = (__bf16)f3.x; r[13] = (__bf16)f3.y; r[14] = (__bf16)f3.z; r[15] = (__bf16)f3.w;
  return r;
}

// CDNA5 async global->LDS copy of 16 bytes per lane (tracked by ASYNCcnt).
#define ASYNC_CP16(ldsAddr, gptr)                                          \
  asm volatile("global_load_async_to_lds_b128 %0, %1, off"                 \
               :: "v"((unsigned)(ldsAddr)),                                \
                  "v"((unsigned long long)(uintptr_t)(gptr))               \
               : "memory")

// ---------------------------------------------------------------------------
// GEMM: C[M,N] = Ab[M,Kpad](bf16) * B[N,Kpad] (+bias)
//   B side: bf16 packed [N,Kpad]  (bF32==0)
//        or fp32 row-major [N,ldbF] (bF32==1; tied lm_head on embedding)
// Block = 8 waves = 2(M) x 4(N); wave tile 32x64 (8 accumulators) ->
// block tile 64(M) x 256(N).  Requires M % 64 == 0, Kpad % 32 == 0.
// A tile (64 rows x 32 K, bf16, 4KB) is staged in LDS via
// global_load_async_to_lds_b128, double-buffered across K-steps, shared by
// all 8 waves.  B fragments are loaded directly from global; each B fragment
// feeds 2 WMMAs (two M sub-tiles).
// Fragment layouts per CDNA5 ISA 7.12.2 (wave32):
//   A/B elem e of lane (h=lane>>4): K = k0 + (e<8 ? h*8+e : 16+h*8+e-8)
//   C/D vgpr r: M = h*8 + r, N = lane&15.
// ---------------------------------------------------------------------------
__global__ __launch_bounds__(256)
void gemm_wmma_bf16(const __bf16* __restrict__ Ab,
                    const __bf16* __restrict__ Bb,
                    const float*  __restrict__ Bf,
                    const float*  __restrict__ bias,
                    float* __restrict__ C,
                    int N, int Kpad, int ldbF, int ldc, int bF32)
{
  __shared__ __align__(16) __bf16 shA[2][64 * 32];   // 2 x 4KB

  const int tid  = threadIdx.x;
  const int lane = tid & 31;
  const int wave = tid >> 5;              // 0..7
  const int wm   = wave & 1;              // M half of block tile
  const int wn   = wave >> 1;             // 0..3 N sub-tile
  const int hh   = lane >> 4;
  const int mr   = lane & 15;

  const int rowBase64 = blockIdx.y * 64;
  const int colBase0  = blockIdx.x * 256 + wn * 64;

  // per-thread async-copy chunk: 64 rows x 32 K bf16 = 256 chunks of 16B
  const int crow = tid >> 2;              // 0..63
  const int cir  = tid & 3;               // 0..3 (8 bf16 each)
  const unsigned ldsBase =
      (unsigned)(uintptr_t)(&shA[0][0]) + (unsigned)(crow * 64 + cir * 16);
  const __bf16* gbase = Ab + (size_t)(rowBase64 + crow) * Kpad + cir * 8;

  v8f acc[4][2];
#pragma unroll
  for (int t = 0; t < 4; ++t)
#pragma unroll
    for (int s = 0; s < 2; ++s)
#pragma unroll
      for (int r = 0; r < 8; ++r) acc[t][s][r] = 0.0f;

  int colc[4];
#pragma unroll
  for (int t = 0; t < 4; ++t) {
    int c = colBase0 + t * 16 + mr;
    colc[t] = (c < N) ? c : (N - 1);      // clamp: loads stay in-bounds
  }
  const __bf16* bpb[4];
  const float*  bpf[4];
#pragma unroll
  for (int t = 0; t < 4; ++t) {
    bpb[t] = Bb + (size_t)colc[t] * Kpad;
    bpf[t] = Bf + (size_t)colc[t] * ldbF;
  }

  const int r0 = wm * 32 + mr;            // local A rows of the two sub-tiles
  const int r1 = wm * 32 + 16 + mr;

  // prologue: stage first A chunk into buffer 0
  ASYNC_CP16(ldsBase, gbase);

  for (int k0 = 0; k0 < Kpad; k0 += 32) {
    const int cur = (k0 >> 5) & 1;
    if (k0 + 32 < Kpad) {
      ASYNC_CP16(ldsBase + (cur ^ 1) * 4096, gbase + k0 + 32);
      asm volatile("s_wait_asynccnt 0x1" ::: "memory");   // current buf done
    } else {
      asm volatile("s_wait_asynccnt 0x0" ::: "memory");
    }
    __syncthreads();

    const __bf16* shp = &shA[cur][0];
    v16bf a0 = cat16(*(const v8bf*)(shp + r0 * 32 + hh * 8),
                     *(const v8bf*)(shp + r0 * 32 + 16 + hh * 8));
    v16bf a1 = cat16(*(const v8bf*)(shp + r1 * 32 + hh * 8),
                     *(const v8bf*)(shp + r1 * 32 + 16 + hh * 8));

    v16bf bfr[4];
    if (!bF32) {
      __builtin_prefetch(bpb[0] + k0 + 32, 0, 0);
#pragma unroll
      for (int t = 0; t < 4; ++t)
        bfr[t] = cat16(*(const v8bf*)(bpb[t] + k0 + hh * 8),
                       *(const v8bf*)(bpb[t] + k0 + 16 + hh * 8));
    } else {
      __builtin_prefetch(bpf[0] + k0 + 32, 0, 0);
#pragma unroll
      for (int t = 0; t < 4; ++t)
        bfr[t] = cvt16(*(const float4*)(bpf[t] + k0 + hh * 8),
                       *(const float4*)(bpf[t] + k0 + hh * 8 + 4),
                       *(const float4*)(bpf[t] + k0 + 16 + hh * 8),
                       *(const float4*)(bpf[t] + k0 + 16 + hh * 8 + 4));
    }

#pragma unroll
    for (int t = 0; t < 4; ++t) {
      acc[t][0] = __builtin_amdgcn_wmma_f32_16x16x32_bf16(
          false, a0, false, bfr[t], (short)0, acc[t][0], false, false);
      acc[t][1] = __builtin_amdgcn_wmma_f32_16x16x32_bf16(
          false, a1, false, bfr[t], (short)0, acc[t][1], false, false);
    }
    __syncthreads();   // all waves done reading shA[cur] before reuse
  }

#pragma unroll
  for (int t = 0; t < 4; ++t) {
    int col = colBase0 + t * 16 + mr;
    if (col >= N) continue;
    float bv = bias ? bias[col] : 0.0f;
#pragma unroll
    for (int s = 0; s < 2; ++s) {
#pragma unroll
      for (int r = 0; r < 8; ++r) {
        int row = rowBase64 + wm * 32 + s * 16 + hh * 8 + r;
        C[(size_t)row * ldc + col] = acc[t][s][r] + bv;
      }
    }
  }
}

// ---------------------------------------------------------------------------
// Pack fp32 activations [rows, lda] -> bf16 [rows, Kpad], zero-padded K..Kpad.
// ---------------------------------------------------------------------------
__global__ void pack_a_bf16(const float* __restrict__ src, __bf16* __restrict__ dst,
                            int rows, int K, int lda, int Kpad)
{
  int idx = blockIdx.x * blockDim.x + threadIdx.x;
  if (idx >= rows * Kpad) return;
  int m = idx / Kpad, k = idx - m * Kpad;
  dst[idx] = (k < K) ? (__bf16)src[(size_t)m * lda + k] : (__bf16)0.0f;
}

// Pack fp32 weights [K, N] -> transposed bf16 [N, Kpad], zero-padded.
__global__ void pack_w_bf16(const float* __restrict__ src, __bf16* __restrict__ dst,
                            int K, int N, int Kpad)
{
  int idx = blockIdx.x * blockDim.x + threadIdx.x;
  if (idx >= N * Kpad) return;
  int n = idx / Kpad, k = idx - n * Kpad;
  dst[idx] = (k < K) ? (__bf16)src[(size_t)k * N + n] : (__bf16)0.0f;
}

// ---------------------------------------------------------------------------
// Embedding gather: x[t, d] = embedding[ids[t], d]
// ---------------------------------------------------------------------------
__global__ void embed_kernel(const int* __restrict__ ids,
                             const float* __restrict__ emb,
                             float* __restrict__ x)
{
  int idx = blockIdx.x * blockDim.x + threadIdx.x;
  if (idx >= NTOK * D_MODEL) return;
  int t = idx / D_MODEL, d = idx - t * D_MODEL;
  x[idx] = emb[(size_t)ids[t] * D_MODEL + d];
}

// ---------------------------------------------------------------------------
// Causal depthwise conv (width 4) + SiLU on xi half; SiLU on res half.
// ---------------------------------------------------------------------------
__global__ void conv_silu_kernel(const float* __restrict__ xr,
                                 const float* __restrict__ cw,
                                 const float* __restrict__ cb,
                                 float* __restrict__ u,
                                 float* __restrict__ res)
{
  int idx = blockIdx.x * blockDim.x + threadIdx.x;
  if (idx >= NTOK * D_INNER) return;
  int t = idx / D_INNER, d = idx - t * D_INNER;
  int b = t / SEQ, l = t - b * SEQ;

  float acc = cb[d];
#pragma unroll
  for (int j = 0; j < D_CONV; ++j) {
    int ls = l + j - (D_CONV - 1);
    if (ls >= 0)
      acc += cw[d * D_CONV + j] *
             xr[((size_t)(b * SEQ + ls)) * (2 * D_INNER) + d];
  }
  u[idx] = silu_f(acc);
  res[idx] = silu_f(xr[(size_t)t * (2 * D_INNER) + D_INNER + d]);
}

// ---------------------------------------------------------------------------
// Selective scan: one thread per (b, d); 16-state in registers; B_t/C_t in LDS.
// ---------------------------------------------------------------------------
__global__ __launch_bounds__(256)
void scan_kernel(const float* __restrict__ dbc,
                 const float* __restrict__ dpre,
                 const float* __restrict__ u,
                 const float* __restrict__ res,
                 const float* __restrict__ A_log,
                 const float* __restrict__ Dp,
                 float* __restrict__ y)
{
  __shared__ float sBC[2 * D_STATE];
  const int d = blockIdx.x * blockDim.x + threadIdx.x;
  const int b = blockIdx.y;

  float negA[D_STATE], s[D_STATE];
#pragma unroll
  for (int n = 0; n < D_STATE; ++n) {
    negA[n] = -expf(A_log[(size_t)d * D_STATE + n]);
    s[n] = 0.0f;
  }
  const float dpv = Dp[d];

  for (int l = 0; l < SEQ; ++l) {
    const size_t t = (size_t)b * SEQ + l;
    __syncthreads();
    if (threadIdx.x < 2 * D_STATE)
      sBC[threadIdx.x] = dbc[t * XPROJ_N + DT_RANK + threadIdx.x];
    __syncthreads();

    const float delta = softplus_f(dpre[t * D_INNER + d]);
    const float uv    = u[t * D_INNER + d];
    float acc = 0.0f;
#pragma unroll
    for (int n = 0; n < D_STATE; ++n) {
      float Aa  = negA[n];
      float dA  = expf(delta * Aa);
      float zoh = (dA - 1.0f) / (Aa + 1e-7f);
      s[n] = dA * s[n] + zoh * sBC[n] * uv;
      acc += s[n] * sBC[D_STATE + n];
    }
    y[t * D_INNER + d] = (acc + uv * dpv) * res[t * D_INNER + d];
  }
}

// ---------------------------------------------------------------------------
// xout = rmsnorm(xin (+ add), w).  One block per row (768 cols, 256 threads).
// ---------------------------------------------------------------------------
__global__ __launch_bounds__(256)
void add_rmsnorm_kernel(const float* __restrict__ add,
                        const float* __restrict__ xin,
                        const float* __restrict__ w,
                        float* __restrict__ xout, int hasAdd)
{
  __shared__ float sh[256];
  const int row = blockIdx.x;
  float v[3];
  float ss = 0.0f;
#pragma unroll
  for (int i = 0; i < 3; ++i) {
    int c = threadIdx.x + i * 256;
    float t = xin[(size_t)row * D_MODEL + c];
    if (hasAdd) t += add[(size_t)row * D_MODEL + c];
    v[i] = t;
    ss += t * t;
  }
  sh[threadIdx.x] = ss;
  __syncthreads();
  for (int off = 128; off > 0; off >>= 1) {
    if (threadIdx.x < off) sh[threadIdx.x] += sh[threadIdx.x + off];
    __syncthreads();
  }
  const float scale = rsqrtf(sh[0] / (float)D_MODEL + 1e-5f);
#pragma unroll
  for (int i = 0; i < 3; ++i) {
    int c = threadIdx.x + i * 256;
    xout[(size_t)row * D_MODEL + c] = v[i] * scale * w[c];
  }
}

// ---------------------------------------------------------------------------
extern "C" void kernel_launch(void* const* d_in, const int* in_sizes, int n_in,
                              void* d_out, int out_size, void* d_ws, size_t ws_size,
                              hipStream_t stream)
{
  const int*   ids     = (const int*)  d_in[0];
  const float* emb     = (const float*)d_in[1];
  const float* in_w    = (const float*)d_in[2];
  const float* in_b    = (const float*)d_in[3];
  const float* conv_w  = (const float*)d_in[4];
  const float* conv_b  = (const float*)d_in[5];
  const float* xproj_w = (const float*)d_in[6];
  const float* dt_w    = (const float*)d_in[7];
  const float* dt_b    = (const float*)d_in[8];
  const float* A_log   = (const float*)d_in[9];
  const float* Dp      = (const float*)d_in[10];
  const float* out_w   = (const float*)d_in[11];
  const float* out_b   = (const float*)d_in[12];
  const float* norm_w  = (const float*)d_in[13];
  const float* normf_w = (const float*)d_in[14];
  float* out = (float*)d_out;

  // ---- fp32 workspace -----------------------------------------------------
  float* ws = (float*)d_ws;
  size_t off = 0;
  float* x     = ws + off; off += (size_t)NTOK * D_MODEL;
  float* xr    = ws + off; off += (size_t)NTOK * 2 * D_INNER;
  float* u     = ws + off; off += (size_t)NTOK * D_INNER;
  float* res   = ws + off; off += (size_t)NTOK * D_INNER;
  float* dbc   = ws + off; off += (size_t)NTOK * XPROJ_N;
  float* dpre  = ws + off; off += (size_t)NTOK * D_INNER;
  float* y     = ws + off; off += (size_t)NTOK * D_INNER;
  float* mixed = ws + off; off += (size_t)NTOK * D_MODEL;
  float* xn    = ws + off; off += (size_t)NTOK * D_MODEL;
  // ---- bf16 workspace (16B-aligned: all prior sizes are multiples of 8) ---
  __bf16* bws  = (__bf16*)(ws + off);
  __bf16* abuf = bws;                               // max NTOK*D_INNER = 3.1M
  __bf16* wbuf = bws + (size_t)NTOK * D_INNER;      // max 3072*768    = 2.36M

  const int P = 256;

  // 1) embedding gather
  embed_kernel<<<(NTOK * D_MODEL + P - 1) / P, P, 0, stream>>>(ids, emb, x);

  for (int layer = 0; layer < N_LAYERS; ++layer) {
    const float* lw_in = in_w    + (size_t)layer * D_MODEL * 2 * D_INNER;
    const float* lb_in = in_b    + (size_t)layer * 2 * D_INNER;
    const float* lcw   = conv_w  + (size_t)layer * D_INNER * D_CONV;
    const float* lcb   = conv_b  + (size_t)layer * D_INNER;
    const float* lxw   = xproj_w + (size_t)layer * D_INNER * XPROJ_N;
    const float* ldtw  = dt_w    + (size_t)layer * DT_RANK * D_INNER;
    const float* ldtb  = dt_b    + (size_t)layer * D_INNER;
    const float* lAlog = A_log   + (size_t)layer * D_INNER * D_STATE;
    const float* lDp   = Dp      + (size_t)layer * D_INNER;
    const float* low   = out_w   + (size_t)layer * D_INNER * D_MODEL;
    const float* lob   = out_b   + (size_t)layer * D_MODEL;
    const float* lnw   = norm_w  + (size_t)layer * D_MODEL;

    // in_proj: xr = x @ in_w + in_b   (2048 x 3072, K=768)
    pack_a_bf16<<<(NTOK * D_MODEL + P - 1) / P, P, 0, stream>>>(
        x, abuf, NTOK, D_MODEL, D_MODEL, D_MODEL);
    pack_w_bf16<<<(2 * D_INNER * D_MODEL + P - 1) / P, P, 0, stream>>>(
        lw_in, wbuf, D_MODEL, 2 * D_INNER, D_MODEL);
    {
      dim3 g((2 * D_INNER + 255) / 256, NTOK / 64);
      gemm_wmma_bf16<<<g, 256, 0, stream>>>(abuf, wbuf, nullptr, lb_in, xr,
          2 * D_INNER, D_MODEL, 0, 2 * D_INNER, 0);
    }
    // conv + silu
    conv_silu_kernel<<<(NTOK * D_INNER + P - 1) / P, P, 0, stream>>>(
        xr, lcw, lcb, u, res);
    // xproj: dbc = u @ xproj_w        (2048 x 80, K=1536)
    pack_a_bf16<<<(NTOK * D_INNER + P - 1) / P, P, 0, stream>>>(
        u, abuf, NTOK, D_INNER, D_INNER, D_INNER);
    pack_w_bf16<<<(XPROJ_N * D_INNER + P - 1) / P, P, 0, stream>>>(
        lxw, wbuf, D_INNER, XPROJ_N, D_INNER);
    {
      dim3 g((XPROJ_N + 255) / 256, NTOK / 64);
      gemm_wmma_bf16<<<g, 256, 0, stream>>>(abuf, wbuf, nullptr, nullptr, dbc,
          XPROJ_N, D_INNER, 0, XPROJ_N, 0);
    }
    // dt_proj: dpre = dt @ dt_w + dt_b  (2048 x 1536, K=48 -> Kpad=64)
    pack_a_bf16<<<(NTOK * DT_PAD + P - 1) / P, P, 0, stream>>>(
        dbc, abuf, NTOK, DT_RANK, XPROJ_N, DT_PAD);
    pack_w_bf16<<<(D_INNER * DT_PAD + P - 1) / P, P, 0, stream>>>(
        ldtw, wbuf, DT_RANK, D_INNER, DT_PAD);
    {
      dim3 g((D_INNER + 255) / 256, NTOK / 64);
      gemm_wmma_bf16<<<g, 256, 0, stream>>>(abuf, wbuf, nullptr, ldtb, dpre,
          D_INNER, DT_PAD, 0, D_INNER, 0);
    }
    // selective scan -> y
    {
      dim3 g(D_INNER / 256, BATCH);
      scan_kernel<<<g, 256, 0, stream>>>(dbc, dpre, u, res, lAlog, lDp, y);
    }
    // out_proj: mixed = y @ out_w + out_b  (2048 x 768, K=1536)
    pack_a_bf16<<<(NTOK * D_INNER + P - 1) / P, P, 0, stream>>>(
        y, abuf, NTOK, D_INNER, D_INNER, D_INNER);
    pack_w_bf16<<<(D_MODEL * D_INNER + P - 1) / P, P, 0, stream>>>(
        low, wbuf, D_INNER, D_MODEL, D_INNER);
    {
      dim3 g((D_MODEL + 255) / 256, NTOK / 64);
      gemm_wmma_bf16<<<g, 256, 0, stream>>>(abuf, wbuf, nullptr, lob, mixed,
          D_MODEL, D_INNER, 0, D_MODEL, 0);
    }
    // residual + rmsnorm -> x
    add_rmsnorm_kernel<<<NTOK, 256, 0, stream>>>(mixed, x, lnw, x, 1);
  }

  // final rmsnorm -> xn
  add_rmsnorm_kernel<<<NTOK, 256, 0, stream>>>(nullptr, x, normf_w, xn, 0);

  // lm_head (tied, fp32-B path on embedding [VOCAB, 768])
  pack_a_bf16<<<(NTOK * D_MODEL + P - 1) / P, P, 0, stream>>>(
      xn, abuf, NTOK, D_MODEL, D_MODEL, D_MODEL);
  {
    dim3 g((VOCAB + 255) / 256, NTOK / 64);
    gemm_wmma_bf16<<<g, 256, 0, stream>>>(abuf, nullptr, emb, nullptr, out,
        VOCAB, D_MODEL, D_MODEL, VOCAB, 1);
  }
}
